// FullAttention_12695923327517
// MI455X (gfx1250) — compile-verified, matrix operand order checked by reading, not run on hardware
//
#include <hip/hip_runtime.h>
#include <hip/hip_bf16.h>

// MI455X / gfx1250, wave32. All GEMMs via v_wmma_f32_16x16x32_bf16; tiles are
// double-buffered in LDS and staged with async global->LDS copies (ASYNCcnt)
// when the toolchain exposes the builtin.
#define B_  4
#define T_  2048
#define C_  1024
#define H_  16
#define HD_ 64
#define M_  (B_*T_)   // 8192 rows

typedef __attribute__((ext_vector_type(16))) __bf16       v16bf;
typedef __attribute__((ext_vector_type(8)))  float        v8f;
typedef __attribute__((ext_vector_type(4)))  unsigned int v4u;
typedef int v4i_ __attribute__((__vector_size__(16)));   // builtin's pointee type

#if defined(__AMDGCN__) && __has_builtin(__builtin_amdgcn_global_load_async_to_lds_b128)
#define ASYNC_LDS 1
#else
#define ASYNC_LDS 0
#endif

static __device__ __forceinline__ unsigned short f32_to_bf16(float f) {
  unsigned int u = __float_as_uint(f);
  u += 0x7fffu + ((u >> 16) & 1u);  // round-to-nearest-even
  return (unsigned short)(u >> 16);
}

// 16B global -> LDS copy; async (no VGPR round trip, ASYNCcnt) if available.
static __device__ __forceinline__ void cp16(const unsigned short* g,
                                            unsigned short* l) {
#if ASYNC_LDS
  __builtin_amdgcn_global_load_async_to_lds_b128(
      (__attribute__((address_space(1))) v4i_*)(unsigned long long)g,
      (__attribute__((address_space(3))) v4i_*)(unsigned)(unsigned long long)l,
      0, 0);
#else
  *(v4u*)l = *(const v4u*)g;
#endif
}
static __device__ __forceinline__ void cp_fence() {
#if ASYNC_LDS
#if __has_builtin(__builtin_amdgcn_s_wait_asynccnt)
  __builtin_amdgcn_s_wait_asynccnt(0);
#else
  asm volatile("s_wait_asynccnt 0x0" ::: "memory");
#endif
#endif
}

union Frag { v4u q[2]; v16bf v; };

// 16x32 bf16 WMMA fragment row for lane-half `hi`. Per ISA 16-bit A layout:
// lanes 0-15 hold K {0..7,16..23}, lanes 16-31 hold K {8..15,24..31}.
static __device__ __forceinline__ v16bf ld_frag(const unsigned short* rowbase,
                                                int koff, int hi) {
  Frag f;
  f.q[0] = *(const v4u*)(rowbase + koff + hi * 8);
  f.q[1] = *(const v4u*)(rowbase + koff + 16 + hi * 8);
  return f.v;
}

static __device__ __forceinline__ v8f wmma_bf16(v16bf a, v16bf b, v8f c) {
  return __builtin_amdgcn_wmma_f32_16x16x32_bf16(false, a, false, b,
                                                 (short)0, c, false, false);
}

// 16-lane reductions (C/D layout: a logical row lives across lanes 0-15 or
// 16-31; xor masks <16 stay inside each half-group).
static __device__ __forceinline__ float redmax16(float v) {
  v = fmaxf(v, __shfl_xor(v, 1, 32));
  v = fmaxf(v, __shfl_xor(v, 2, 32));
  v = fmaxf(v, __shfl_xor(v, 4, 32));
  v = fmaxf(v, __shfl_xor(v, 8, 32));
  return v;
}
static __device__ __forceinline__ float redsum16(float v) {
  v += __shfl_xor(v, 1, 32);
  v += __shfl_xor(v, 2, 32);
  v += __shfl_xor(v, 4, 32);
  v += __shfl_xor(v, 8, 32);
  return v;
}

// ---------------------------------------------------------------- convert
__global__ __launch_bounds__(256)
void cvt_f32_bf16(const float* __restrict__ src,
                  unsigned short* __restrict__ dst, int n) {
  int i = blockIdx.x * 256 + threadIdx.x;
  if (i < n) dst[i] = f32_to_bf16(src[i]);
}

// ---------------------------------------------------------------- GEMM
// out[m,n] = sum_k A[m,k]*W[n,k] + bias[n].  M=8192, N=1024, K=1024.
// Block 256 thr = 8 waves; block tile 128(M) x 128(N); wave tile 32x64.
// Double-buffered LDS, one barrier per 32-K step, async staging overlapped
// with WMMA. MODE 0: bf16 [B,H,T,hd] (QKV). MODE 1: fp32 [M,N] (out proj).
template <int MODE>
__global__ __launch_bounds__(256)
void gemm_bf16k(const unsigned short* __restrict__ A,
                const unsigned short* __restrict__ W,
                const float* __restrict__ bias,
                void* __restrict__ outp) {
  __shared__ unsigned short bA[2][128 * 40];  // 32 bf16 + pad (16B rows)
  __shared__ unsigned short bB[2][128 * 40];
  const int tid = threadIdx.x;
  const int wave = tid >> 5, lane = tid & 31;
  const int hi = lane >> 4, ln = lane & 15;
  const int wm = wave >> 1;  // 0..3 : 32-row slice
  const int wn = wave & 1;   // 0..1 : 64-col slice
  const int m0 = blockIdx.x * 128;
  const int n0 = blockIdx.y * 128;

  const v8f vz = {0.f, 0.f, 0.f, 0.f, 0.f, 0.f, 0.f, 0.f};
  v8f acc[2][4];
#pragma unroll
  for (int a2 = 0; a2 < 2; ++a2)
#pragma unroll
    for (int t = 0; t < 4; ++t) acc[a2][t] = vz;

  // per-thread staging map: 512 16B chunks per 128x32 tile, 2 per thread
  const int c0 = tid * 2;
  auto stage = [&](int kk, int pp) {
#pragma unroll
    for (int i = 0; i < 2; ++i) {
      int c = c0 + i;
      int row = c >> 2, cs = c & 3;
      cp16(A + (size_t)(m0 + row) * C_ + kk + cs * 8,
           &bA[pp][row * 40 + cs * 8]);
      cp16(W + (size_t)(n0 + row) * C_ + kk + cs * 8,
           &bB[pp][row * 40 + cs * 8]);
    }
  };

  stage(0, 0);
  cp_fence();
  __syncthreads();

  int p = 0;
  for (int kk = 0; kk < C_; kk += 32) {
    if (kk + 32 < C_) stage(kk + 32, p ^ 1);  // overlap with compute below

    v16bf af0 = ld_frag(&bA[p][(wm * 32 + ln) * 40], 0, hi);
    v16bf af1 = ld_frag(&bA[p][(wm * 32 + 16 + ln) * 40], 0, hi);
#pragma unroll
    for (int t = 0; t < 4; ++t) {
      v16bf bf = ld_frag(&bB[p][(wn * 64 + t * 16 + ln) * 40], 0, hi);
      acc[0][t] = wmma_bf16(af0, bf, acc[0][t]);
      acc[1][t] = wmma_bf16(af1, bf, acc[1][t]);
    }
    cp_fence();
    __syncthreads();
    p ^= 1;
  }

#pragma unroll
  for (int a2 = 0; a2 < 2; ++a2)
#pragma unroll
    for (int t = 0; t < 4; ++t) {
      int n = n0 + wn * 64 + t * 16 + ln;
      float bv = bias[n];
#pragma unroll
      for (int r = 0; r < 8; ++r) {
        int m = m0 + wm * 32 + a2 * 16 + r + 8 * hi;
        float val = acc[a2][t][r] + bv;
        if (MODE == 0) {
          unsigned short* out = (unsigned short*)outp;
          int b = m >> 11, tt = m & (T_ - 1);
          int h = n >> 6, d = n & (HD_ - 1);
          out[((size_t)((b * H_ + h) * T_ + tt)) * HD_ + d] = f32_to_bf16(val);
        } else {
          float* out = (float*)outp;
          out[(size_t)m * C_ + n] = val;
        }
      }
    }
}

// ---------------------------------------------------------------- flash fwd
// One WG per (b, h, 64 q rows). 4 waves x 16 q rows. Online softmax over 32
// k-blocks of 64; K/V tiles double-buffered in LDS (K async-staged, V staged
// transposed), one barrier per k-block. Writes per-head output (bf16,
// re-merged [B*T, C]) and per-row (m, l) stats for the att_mean pass.
__global__ __launch_bounds__(128)
void flash_fwd(const unsigned short* __restrict__ Q,
               const unsigned short* __restrict__ K,
               const unsigned short* __restrict__ V,
               unsigned short* __restrict__ Yc,
               float* __restrict__ ml) {
  __shared__ unsigned short Kb[2][64 * 72];     // K block, [k][d], padded
  __shared__ unsigned short Vt[2][64 * 72];     // V block transposed, [d][k]
  __shared__ unsigned short Pb[4 * 16 * 72];    // per-wave P staging
  const int tid = threadIdx.x;
  const int wave = tid >> 5, lane = tid & 31;
  const int hi = lane >> 4, ln = lane & 15;
  const int qb = blockIdx.x, h = blockIdx.y, b = blockIdx.z;
  const size_t headBase = ((size_t)(b * H_ + h)) * T_ * HD_;
  const float scale = 0.125f;  // 1/sqrt(64)
  const int NB = T_ / 64;

  // Q fragments for this wave's 16 rows (hd=64 -> 2 k-steps of 32)
  Frag qf[2];
  {
    const unsigned short* qrow =
        Q + headBase + (size_t)(qb * 64 + wave * 16 + ln) * HD_;
#pragma unroll
    for (int s = 0; s < 2; ++s) {
      qf[s].q[0] = *(const v4u*)(qrow + s * 32 + hi * 8);
      qf[s].q[1] = *(const v4u*)(qrow + s * 32 + 16 + hi * 8);
    }
  }

  const v8f vz = {0.f, 0.f, 0.f, 0.f, 0.f, 0.f, 0.f, 0.f};
  v8f acc[4];
#pragma unroll
  for (int t = 0; t < 4; ++t) acc[t] = vz;
  float mrow[8], lrow[8];
#pragma unroll
  for (int r = 0; r < 8; ++r) { mrow[r] = -1e30f; lrow[r] = 0.f; }

  auto stageKV = [&](int kb, int pp) {
#pragma unroll
    for (int i = 0; i < 4; ++i) {
      int c = tid + i * 128;
      int row = c >> 3, cs = c & 7;
      cp16(K + headBase + (size_t)(kb * 64 + row) * HD_ + cs * 8,
           &Kb[pp][row * 72 + cs * 8]);
    }
#pragma unroll
    for (int i = 0; i < 4; ++i) {  // V needs an element transpose: sync path
      int c = tid + i * 128;
      int row = c >> 3, cs = c & 7;
      union { v4u q; unsigned short s[8]; } u;
      u.q = *(const v4u*)(V + headBase + (size_t)(kb * 64 + row) * HD_ + cs * 8);
#pragma unroll
      for (int j = 0; j < 8; ++j) Vt[pp][(cs * 8 + j) * 72 + row] = u.s[j];
    }
  };

  stageKV(0, 0);
  cp_fence();
  __syncthreads();

  int p = 0;
  for (int kb = 0; kb < NB; ++kb) {
    if (kb + 1 < NB) stageKV(kb + 1, p ^ 1);  // overlapped with compute
    if (kb + 2 < NB) {                        // L2 prefetch two blocks ahead
      __builtin_prefetch(K + headBase + (size_t)((kb + 2) * 64) * HD_ + tid * 32, 0, 1);
      __builtin_prefetch(V + headBase + (size_t)((kb + 2) * 64) * HD_ + tid * 32, 0, 1);
    }

    // S(16x64) = Q(16x64) * K^T
    v8f s[4];
#pragma unroll
    for (int t = 0; t < 4; ++t) s[t] = vz;
#pragma unroll
    for (int st = 0; st < 2; ++st) {
#pragma unroll
      for (int t = 0; t < 4; ++t) {
        v16bf kf = ld_frag(&Kb[p][(t * 16 + ln) * 72], st * 32, hi);
        s[t] = wmma_bf16(qf[st].v, kf, s[t]);
      }
    }
#pragma unroll
    for (int t = 0; t < 4; ++t)
#pragma unroll
      for (int r = 0; r < 8; ++r) s[t][r] *= scale;

    // online softmax stats
    float corr[8];
#pragma unroll
    for (int r = 0; r < 8; ++r) {
      float mx = fmaxf(fmaxf(s[0][r], s[1][r]), fmaxf(s[2][r], s[3][r]));
      mx = redmax16(mx);
      float mnew = fmaxf(mrow[r], mx);
      corr[r] = __expf(mrow[r] - mnew);
      mrow[r] = mnew;
    }
#pragma unroll
    for (int t = 0; t < 4; ++t)
#pragma unroll
      for (int r = 0; r < 8; ++r) s[t][r] = __expf(s[t][r] - mrow[r]);
#pragma unroll
    for (int r = 0; r < 8; ++r) {
      float sm = s[0][r] + s[1][r] + s[2][r] + s[3][r];
      sm = redsum16(sm);
      lrow[r] = lrow[r] * corr[r] + sm;
    }
#pragma unroll
    for (int t = 0; t < 4; ++t)
#pragma unroll
      for (int r = 0; r < 8; ++r) acc[t][r] *= corr[r];

    // round-trip P through per-wave LDS: C/D layout -> A-fragment layout
    // (wave-private region; per-wave LDS ordering, no barrier needed)
    unsigned short* pw = &Pb[wave * 16 * 72];
#pragma unroll
    for (int t = 0; t < 4; ++t)
#pragma unroll
      for (int r = 0; r < 8; ++r)
        pw[(r + 8 * hi) * 72 + t * 16 + ln] = f32_to_bf16(s[t][r]);

    // O += P(16x64) * V(64x64)
#pragma unroll
    for (int st = 0; st < 2; ++st) {
      v16bf pf = ld_frag(&pw[ln * 72], st * 32, hi);
#pragma unroll
      for (int t = 0; t < 4; ++t) {
        v16bf vf = ld_frag(&Vt[p][(t * 16 + ln) * 72], st * 32, hi);
        acc[t] = wmma_bf16(pf, vf, acc[t]);
      }
    }
    cp_fence();
    __syncthreads();
    p ^= 1;
  }

  // normalize, write merged [B*T, C] bf16 (input of output projection GEMM)
#pragma unroll
  for (int t = 0; t < 4; ++t)
#pragma unroll
    for (int r = 0; r < 8; ++r) {
      int q = qb * 64 + wave * 16 + r + 8 * hi;
      int d = t * 16 + ln;
      float val = acc[t][r] / lrow[r];
      Yc[((size_t)(b * T_ + q)) * C_ + h * HD_ + d] = f32_to_bf16(val);
    }
  if (ln == 0) {  // lanes 0 and 16 cover all 16 rows
#pragma unroll
    for (int r = 0; r < 8; ++r) {
      int q = qb * 64 + wave * 16 + r + 8 * hi;
      size_t idx = (((size_t)(b * H_ + h)) * T_ + q) * 2;
      ml[idx] = mrow[r];
      ml[idx + 1] = lrow[r];
    }
  }
}

// ---------------------------------------------------------------- att_mean
// One WG per (b, 64x64 tile of [T,T]); loops all 16 heads internally,
// recomputes S via WMMA (double-buffered K tile), applies stored softmax
// stats, accumulates the head-mean.
__global__ __launch_bounds__(128)
void att_mean_k(const unsigned short* __restrict__ Q,
                const unsigned short* __restrict__ K,
                const float* __restrict__ ml,
                float* __restrict__ att) {
  __shared__ unsigned short Kb[2][64 * 72];
  const int tid = threadIdx.x;
  const int wave = tid >> 5, lane = tid & 31;
  const int hi = lane >> 4, ln = lane & 15;
  const int kb = blockIdx.x, qb = blockIdx.y, b = blockIdx.z;
  const float scale = 0.125f;

  const v8f vz = {0.f, 0.f, 0.f, 0.f, 0.f, 0.f, 0.f, 0.f};
  v8f acc[4];
#pragma unroll
  for (int t = 0; t < 4; ++t) acc[t] = vz;

  auto stageK = [&](int h, int pp) {
    const size_t hb = ((size_t)(b * H_ + h)) * T_ * HD_;
#pragma unroll
    for (int i = 0; i < 4; ++i) {
      int c = tid + i * 128;
      int row = c >> 3, cs = c & 7;
      cp16(K + hb + (size_t)(kb * 64 + row) * HD_ + cs * 8,
           &Kb[pp][row * 72 + cs * 8]);
    }
  };

  stageK(0, 0);
  cp_fence();
  __syncthreads();

  int p = 0;
  for (int h = 0; h < H_; ++h) {
    if (h + 1 < H_) stageK(h + 1, p ^ 1);

    const size_t headBase = ((size_t)(b * H_ + h)) * T_ * HD_;
    Frag qf[2];
    const unsigned short* qrow =
        Q + headBase + (size_t)(qb * 64 + wave * 16 + ln) * HD_;
#pragma unroll
    for (int s = 0; s < 2; ++s) {
      qf[s].q[0] = *(const v4u*)(qrow + s * 32 + hi * 8);
      qf[s].q[1] = *(const v4u*)(qrow + s * 32 + 16 + hi * 8);
    }

    v8f s[4];
#pragma unroll
    for (int t = 0; t < 4; ++t) s[t] = vz;
#pragma unroll
    for (int st = 0; st < 2; ++st) {
#pragma unroll
      for (int t = 0; t < 4; ++t) {
        v16bf kf = ld_frag(&Kb[p][(t * 16 + ln) * 72], st * 32, hi);
        s[t] = wmma_bf16(qf[st].v, kf, s[t]);
      }
    }

    float mr[8], lr[8];
#pragma unroll
    for (int r = 0; r < 8; ++r) {
      int q = qb * 64 + wave * 16 + r + 8 * hi;
      size_t idx = (((size_t)(b * H_ + h)) * T_ + q) * 2;
      mr[r] = ml[idx];
      lr[r] = ml[idx + 1];
    }
#pragma unroll
    for (int t = 0; t < 4; ++t)
#pragma unroll
      for (int r = 0; r < 8; ++r)
        acc[t][r] += __expf(s[t][r] * scale - mr[r]) / lr[r] * (1.0f / H_);

    cp_fence();
    __syncthreads();
    p ^= 1;
  }

#pragma unroll
  for (int t = 0; t < 4; ++t)
#pragma unroll
    for (int r = 0; r < 8; ++r) {
      int q = qb * 64 + wave * 16 + r + 8 * hi;
      int k = kb * 64 + t * 16 + ln;
      att[((size_t)(b * T_ + q)) * T_ + k] = acc[t][r];
    }
}

// ---------------------------------------------------------------- launch
extern "C" void kernel_launch(void* const* d_in, const int* in_sizes, int n_in,
                              void* d_out, int out_size, void* d_ws, size_t ws_size,
                              hipStream_t stream) {
  (void)in_sizes; (void)n_in; (void)out_size; (void)ws_size;
  const float* x  = (const float*)d_in[0];
  const float* Wq = (const float*)d_in[1];
  const float* bq = (const float*)d_in[2];
  const float* Wk = (const float*)d_in[3];
  const float* bk = (const float*)d_in[4];
  const float* Wv = (const float*)d_in[5];
  const float* bv = (const float*)d_in[6];
  const float* Wp = (const float*)d_in[7];
  const float* bp = (const float*)d_in[8];

  // Workspace layout (~90 MiB total)
  char* ws = (char*)d_ws;
  size_t off = 0;
  auto us = [&](size_t n) {
    unsigned short* p = (unsigned short*)(ws + off);
    off += n * sizeof(unsigned short);
    return p;
  };
  unsigned short* xb  = us((size_t)M_ * C_);
  unsigned short* wqb = us((size_t)C_ * C_);
  unsigned short* wkb = us((size_t)C_ * C_);
  unsigned short* wvb = us((size_t)C_ * C_);
  unsigned short* wpb = us((size_t)C_ * C_);
  unsigned short* qb  = us((size_t)M_ * C_);
  unsigned short* kb  = us((size_t)M_ * C_);
  unsigned short* vb  = us((size_t)M_ * C_);
  unsigned short* yc  = us((size_t)M_ * C_);
  float* ml = (float*)(ws + off);
  off += (size_t)B_ * H_ * T_ * 2 * sizeof(float);

  // 1) fp32 -> bf16
  cvt_f32_bf16<<<dim3((M_ * C_ + 255) / 256), 256, 0, stream>>>(x, xb, M_ * C_);
  cvt_f32_bf16<<<dim3((C_ * C_ + 255) / 256), 256, 0, stream>>>(Wq, wqb, C_ * C_);
  cvt_f32_bf16<<<dim3((C_ * C_ + 255) / 256), 256, 0, stream>>>(Wk, wkb, C_ * C_);
  cvt_f32_bf16<<<dim3((C_ * C_ + 255) / 256), 256, 0, stream>>>(Wv, wvb, C_ * C_);
  cvt_f32_bf16<<<dim3((C_ * C_ + 255) / 256), 256, 0, stream>>>(Wp, wpb, C_ * C_);

  // 2) QKV projections -> [B,H,T,hd] bf16
  dim3 gg(M_ / 128, C_ / 128);
  gemm_bf16k<0><<<gg, 256, 0, stream>>>(xb, wqb, bq, qb);
  gemm_bf16k<0><<<gg, 256, 0, stream>>>(xb, wkb, bk, kb);
  gemm_bf16k<0><<<gg, 256, 0, stream>>>(xb, wvb, bv, vb);

  // 3) flash attention pass 1 (y heads + softmax stats)
  flash_fwd<<<dim3(T_ / 64, H_, B_), 128, 0, stream>>>(qb, kb, vb, yc, ml);

  // 4) output projection -> fp32 y region of d_out
  gemm_bf16k<1><<<gg, 256, 0, stream>>>(yc, wpb, bp, d_out);

  // 5) att_mean pass 2 -> second region of d_out
  att_mean_k<<<dim3(T_ / 64, T_ / 64, B_), 128, 0, stream>>>(
      qb, kb, ml, (float*)d_out + (size_t)M_ * C_);
}
